// UnconditionalDiffWave_21294447854238
// MI455X (gfx1250) — matrix operand order repaired and checked by reading, not compiled, not run on hardware
//
#include <hip/hip_runtime.h>
#include <math.h>

#define B_      16
#define L_      32768
#define C_      64
#define LAYERS_ 30

#define XSTR 72          // padded row stride (halfwords) for bf16 tiles: 144B, 16B-aligned
#define VBSTR 132        // padded row stride (floats) for pos-major f32 tile: 528B, 16B-aligned

typedef unsigned short u16;
typedef __attribute__((ext_vector_type(16))) __bf16 v16bf;
typedef __attribute__((ext_vector_type(8)))  __bf16 v8bf;
typedef __attribute__((ext_vector_type(8)))  float  v8f;
typedef __attribute__((ext_vector_type(4)))  float  v4f;
typedef __attribute__((ext_vector_type(4)))  u16    v4u16;

// ---- helpers -------------------------------------------------------------

__device__ __forceinline__ u16 f2bf(float f) {            // f32 -> bf16, RNE
    unsigned u = __builtin_bit_cast(unsigned, f);
    u += 0x7FFFu + ((u >> 16) & 1u);
    return (u16)(u >> 16);
}
__device__ __forceinline__ float bf2f(u16 h) {
    unsigned u = ((unsigned)h) << 16;
    return __builtin_bit_cast(float, u);
}
// fragment = concat of two 8-element bf16 runs (each 16B-aligned load)
__device__ __forceinline__ v16bf ld_frag16(const u16* p0, const u16* p1) {
    v8bf a = *(const v8bf*)p0;
    v8bf b = *(const v8bf*)p1;
    return __builtin_shufflevector(a, b, 0,1,2,3,4,5,6,7,8,9,10,11,12,13,14,15);
}
__device__ __forceinline__ v8f wmma_bf16(v16bf a, v16bf b, v8f c) {
    return __builtin_amdgcn_wmma_f32_16x16x32_bf16(false, a, false, b, (short)0, c, false, false);
}

// ---- diffusion embedding MLP + per-layer dp vectors (tiny) ---------------

__global__ void embed_dp_kernel(const int* __restrict__ step, const float* __restrict__ table,
                                const float* __restrict__ W1, const float* __restrict__ b1,
                                const float* __restrict__ W2, const float* __restrict__ b2,
                                const float* __restrict__ diffW, const float* __restrict__ diffb,
                                float* __restrict__ dp) {
    __shared__ float t0[128], t1[512], t2[512];
    int b = blockIdx.x, tid = threadIdx.x;
    int s = step[b];
    if (tid < 128) t0[tid] = table[s * 128 + tid];
    __syncthreads();
    for (int j = tid; j < 512; j += 256) {
        float a = b1[j];
        const float* w = W1 + j * 128;
        for (int k = 0; k < 128; ++k) a += t0[k] * w[k];
        t1[j] = a / (1.f + expf(-a));                      // silu
    }
    __syncthreads();
    for (int j = tid; j < 512; j += 256) {
        float a = b2[j];
        const float* w = W2 + j * 512;
        for (int k = 0; k < 512; ++k) a += t1[k] * w[k];
        t2[j] = a / (1.f + expf(-a));
    }
    __syncthreads();
    for (int idx = tid; idx < LAYERS_ * C_; idx += 256) {
        int i = idx / C_, c = idx % C_;
        const float* w = diffW + (size_t)(i * C_ + c) * 512;
        float a = diffb[i * C_ + c];
        for (int k = 0; k < 512; ++k) a += t2[k] * w[k];
        dp[(i * B_ + b) * C_ + c] = a;
    }
}

// ---- one-shot weight conversion to bf16 (conv reordered to [i][tap][o][c]) --

__global__ void prep_weights_kernel(const float* __restrict__ dconvW, const float* __restrict__ outW,
                                    const float* __restrict__ Wo1,
                                    u16* __restrict__ convWb, u16* __restrict__ outWb,
                                    u16* __restrict__ Wo1b) {
    int idx = blockIdx.x * 256 + threadIdx.x;
    const int NC = LAYERS_ * 3 * 128 * C_;
    const int NO = LAYERS_ * 128 * C_;
    if (idx < NC) {
        int c = idx % C_;
        int t = idx / C_;
        int o = t % 128; t /= 128;
        int tap = t % 3; int i = t / 3;
        convWb[idx] = f2bf(dconvW[((size_t)(i * 128 + o) * C_ + c) * 3 + tap]);
    } else if (idx < NC + NO) {
        int k = idx - NC;
        outWb[k] = f2bf(outW[k]);
    } else if (idx < NC + NO + C_ * C_) {
        int k = idx - NC - NO;
        Wo1b[k] = f2bf(Wo1[k]);
    }
}

// ---- input projection: x = relu(audio*Win + bin), zero skip --------------

__global__ void input_proj_kernel(const float* __restrict__ audio, const float* __restrict__ Win,
                                  const float* __restrict__ bin, float* __restrict__ x,
                                  float* __restrict__ skip) {
    size_t idx = (size_t)blockIdx.x * 256 + threadIdx.x;     // layout [B][L][C]
    int c = (int)(idx & (C_ - 1));
    size_t bl = idx >> 6;
    float v = audio[bl] * Win[c] + bin[c];
    x[idx] = v > 0.f ? v : 0.f;
    skip[idx] = 0.f;
}

// ---- fused residual layer: dilated conv -> gate -> 1x1 -> residual/skip --
// grid = B * (L/64) blocks of 256 threads (8 wave32). Tile: 64 pos x 128 out ch.

__global__ __launch_bounds__(256)
void layer_kernel(const float* __restrict__ xin, float* __restrict__ xout,
                  float* __restrict__ skip, const float* __restrict__ dp,
                  const u16* __restrict__ wConv,   // [3][128][64] bf16
                  const u16* __restrict__ wOut,    // [128][64] bf16
                  const float* __restrict__ cbias, const float* __restrict__ obias,
                  int d) {
    __shared__ __align__(16) u16   xs[64 * XSTR];     // staged (x+dp) tile, one tap at a time
    __shared__ __align__(16) u16   ys[64 * XSTR];     // gated activation, [pos][c]
    __shared__ __align__(16) float vb[64 * VBSTR];    // pre-gate / z buffer, pos-major [pos][o]
    __shared__ __align__(16) float dps[64];
    __shared__ float cbs[128], obs[128];

    const int tid = threadIdx.x;
    const int blk = blockIdx.x;
    const int b  = blk >> 9;                 // 512 tiles per batch
    const int l0 = (blk & 511) << 6;

    if (tid < 64)  dps[tid] = dp[b * C_ + tid];
    if (tid < 128) { cbs[tid] = cbias[tid]; obs[tid] = obias[tid]; }
    __syncthreads();

    // vectorized staging mapping: each thread owns 4 channels at fixed c4,
    // positions j = jb + 16k (k = 0..3)
    const int c4 = (tid & 15) * 4;
    const int jb = tid >> 4;
    const v4f dp4 = *(const v4f*)&dps[c4];

    const int lane  = tid & 31;
    const int w     = tid >> 5;              // wave id: output rows 16w..16w+15
    const int mrow  = lane & 15;
    const int hi    = lane >> 4;
    const int koffA = hi * 8;                // A-frag K offset per ISA 16-bit A layout
    const int koffB = hi * 8;                // B-frag K offset, same table order as A
    const int ncol  = lane & 15;

    v8f acc[4];
    #pragma unroll
    for (int nt = 0; nt < 4; ++nt)
        #pragma unroll
        for (int j = 0; j < 8; ++j) acc[nt][j] = 0.f;

    // GEMM1: v[o][l] = sum_tap sum_c wConv[tap][o][c] * (x+dp)[c][l + (tap-1)*d]
    #pragma unroll
    for (int tap = 0; tap < 3; ++tap) {
        const long lmin = (long)l0 + (long)(tap - 1) * d;
        if (lmin >= 0 && lmin + 63 < L_) {
            // fast path: whole tap tile in range, no per-element checks
            const float* base = xin + ((size_t)b * L_ + (size_t)lmin) * C_ + c4;
            #pragma unroll
            for (int k = 0; k < 4; ++k) {
                int j = jb + k * 16;
                v4f xv = *(const v4f*)(base + (size_t)j * C_);
                v4u16 o;
                #pragma unroll
                for (int q = 0; q < 4; ++q) o[q] = f2bf(xv[q] + dp4[q]);
                *(v4u16*)&xs[j * XSTR + c4] = o;
            }
        } else {
            #pragma unroll
            for (int k = 0; k < 4; ++k) {
                int j = jb + k * 16;
                long l = lmin + j;
                v4u16 o = {0, 0, 0, 0};
                if (l >= 0 && l < L_) {
                    v4f xv = *(const v4f*)(xin + ((size_t)b * L_ + (size_t)l) * C_ + c4);
                    #pragma unroll
                    for (int q = 0; q < 4; ++q) o[q] = f2bf(xv[q] + dp4[q]);
                }
                *(v4u16*)&xs[j * XSTR + c4] = o;
            }
        }
        __syncthreads();
        #pragma unroll
        for (int kc = 0; kc < 2; ++kc) {
            const u16* wp = wConv + (size_t)(tap * 128 + w * 16 + mrow) * C_ + kc * 32 + koffA;
            v16bf a = ld_frag16(wp, wp + 16);
            #pragma unroll
            for (int nt = 0; nt < 4; ++nt) {
                const u16* bp = xs + (nt * 16 + ncol) * XSTR + kc * 32 + koffB;
                v16bf bb = ld_frag16(bp, bp + 16);
                acc[nt] = wmma_bf16(a, bb, acc[nt]);
            }
        }
        __syncthreads();                                   // xs reused next tap
    }

    // scatter pre-gate values (+conv bias) to pos-major LDS: vb[n][o]
    // D-frag: lane holds (o = 16w + 8*hi + j, n = 16*nt + ncol)
    #pragma unroll
    for (int nt = 0; nt < 4; ++nt)
        #pragma unroll
        for (int j = 0; j < 8; ++j) {
            int o = w * 16 + hi * 8 + j;
            vb[(nt * 16 + ncol) * VBSTR + o] = acc[nt][j] + cbs[o];
        }
    __syncthreads();

    // gated activation: y = tanh(v[:64]) * sigmoid(v[64:]) -> bf16 [pos][c]
    // vector LDS reads: v4f over channels at fixed position
    {
        const int jg = tid & 63;
        const int cb = (tid >> 6) * 16;
        const float* vrow = &vb[jg * VBSTR];
        #pragma unroll
        for (int k4 = 0; k4 < 4; ++k4) {
            v4f av = *(const v4f*)(vrow + cb + k4 * 4);
            v4f gv = *(const v4f*)(vrow + 64 + cb + k4 * 4);
            v4u16 o;
            #pragma unroll
            for (int q = 0; q < 4; ++q) {
                float th = 1.f - 2.f / (__expf(2.f * av[q]) + 1.f);   // tanh
                float sg = 1.f / (1.f + __expf(-gv[q]));              // sigmoid
                o[q] = f2bf(th * sg);
            }
            *(v4u16*)&ys[jg * XSTR + cb + k4 * 4] = o;
        }
    }
    __syncthreads();

    // GEMM2: z[o][l] = wOut[o][c] * y[c][l]
    v8f acc2[4];
    #pragma unroll
    for (int nt = 0; nt < 4; ++nt)
        #pragma unroll
        for (int j = 0; j < 8; ++j) acc2[nt][j] = 0.f;
    #pragma unroll
    for (int kc = 0; kc < 2; ++kc) {
        const u16* wp = wOut + (size_t)(w * 16 + mrow) * C_ + kc * 32 + koffA;
        v16bf a = ld_frag16(wp, wp + 16);
        #pragma unroll
        for (int nt = 0; nt < 4; ++nt) {
            const u16* bp = ys + (nt * 16 + ncol) * XSTR + kc * 32 + koffB;
            v16bf bb = ld_frag16(bp, bp + 16);
            acc2[nt] = wmma_bf16(a, bb, acc2[nt]);
        }
    }
    #pragma unroll
    for (int nt = 0; nt < 4; ++nt)
        #pragma unroll
        for (int j = 0; j < 8; ++j) {
            int o = w * 16 + hi * 8 + j;
            vb[(nt * 16 + ncol) * VBSTR + o] = acc2[nt][j] + obs[o];
        }
    __syncthreads();

    // residual + skip, float4 coalesced global I/O, vector LDS reads
    {
        const float inv_sqrt2 = 0.70710678118654752f;
        const size_t gbase = ((size_t)b * L_ + l0) * C_;
        const float* xi = xin + gbase;
        float* xo = xout + gbase;
        float* sk = skip + gbase;
        #pragma unroll
        for (int k = 0; k < 4; ++k) {
            int j = jb + k * 16;
            v4f xv = *(const v4f*)(xi + (size_t)j * C_ + c4);
            v4f sv = *(const v4f*)(sk + (size_t)j * C_ + c4);
            v4f z1 = *(const v4f*)&vb[j * VBSTR + c4];
            v4f z2 = *(const v4f*)&vb[j * VBSTR + 64 + c4];
            *(v4f*)(xo + (size_t)j * C_ + c4) = (xv + z1) * inv_sqrt2;
            *(v4f*)(sk + (size_t)j * C_ + c4) = sv + z2;
        }
    }
}

// ---- output head: relu(Wo1 * skip/sqrt(30) + bo1) -> Wo2 dot -------------

__global__ __launch_bounds__(256)
void final_kernel(const float* __restrict__ skip, const u16* __restrict__ Wo1b,
                  const float* __restrict__ bo1, const float* __restrict__ Wo2,
                  const float* __restrict__ bo2, float* __restrict__ out) {
    __shared__ __align__(16) u16 hs[64 * XSTR];
    __shared__ __align__(16) u16 h2s[64 * XSTR];
    __shared__ float bo1s[64];
    const int tid = threadIdx.x;
    const int blk = blockIdx.x;
    const int b  = blk >> 9;
    const int l0 = (blk & 511) << 6;
    const float rs = 0.18257418583505537f;   // 1/sqrt(30)

    if (tid < 64) bo1s[tid] = bo1[tid];
    {
        const int c4 = (tid & 15) * 4;
        const int jb = tid >> 4;
        const float* base = skip + ((size_t)b * L_ + l0) * C_ + c4;
        #pragma unroll
        for (int k = 0; k < 4; ++k) {
            int j = jb + k * 16;
            v4f xv = *(const v4f*)(base + (size_t)j * C_);
            v4u16 o;
            #pragma unroll
            for (int q = 0; q < 4; ++q) o[q] = f2bf(xv[q] * rs);
            *(v4u16*)&hs[j * XSTR + c4] = o;
        }
    }
    __syncthreads();

    const int lane = tid & 31, w = tid >> 5;
    const int mrow = lane & 15, hi = lane >> 4, ncol = lane & 15;
    #pragma unroll
    for (int t = 0; t < 2; ++t) {
        int tt = w * 2 + t;
        int mt = tt >> 2, nt = tt & 3;
        v8f acc;
        #pragma unroll
        for (int j = 0; j < 8; ++j) acc[j] = 0.f;
        #pragma unroll
        for (int kc = 0; kc < 2; ++kc) {
            const u16* wp = Wo1b + (size_t)(mt * 16 + mrow) * C_ + kc * 32 + hi * 8;
            v16bf a = ld_frag16(wp, wp + 16);
            const u16* bp = hs + (nt * 16 + ncol) * XSTR + kc * 32 + hi * 8;
            v16bf bb = ld_frag16(bp, bp + 16);
            acc = wmma_bf16(a, bb, acc);
        }
        #pragma unroll
        for (int j = 0; j < 8; ++j) {
            int m = mt * 16 + hi * 8 + j;
            float v = acc[j] + bo1s[m];
            h2s[(nt * 16 + ncol) * XSTR + m] = f2bf(v > 0.f ? v : 0.f);
        }
    }
    __syncthreads();

    if (tid < 64) {
        float s = bo2[0];
        const u16* hp = h2s + tid * XSTR;
        for (int c = 0; c < C_; ++c) s += Wo2[c] * bf2f(hp[c]);
        out[(size_t)b * L_ + l0 + tid] = s;
    }
}

// ---- host orchestration --------------------------------------------------

extern "C" void kernel_launch(void* const* d_in, const int* in_sizes, int n_in,
                              void* d_out, int out_size, void* d_ws, size_t ws_size,
                              hipStream_t stream) {
    (void)in_sizes; (void)n_in; (void)out_size; (void)ws_size;
    const float* audio  = (const float*)d_in[0];
    const int*   step   = (const int*)  d_in[1];
    const float* table  = (const float*)d_in[2];
    const float* W1     = (const float*)d_in[3];
    const float* b1     = (const float*)d_in[4];
    const float* W2     = (const float*)d_in[5];
    const float* b2     = (const float*)d_in[6];
    const float* Win    = (const float*)d_in[7];
    const float* bin    = (const float*)d_in[8];
    const float* diffW  = (const float*)d_in[9];
    const float* diffb  = (const float*)d_in[10];
    const float* dconvW = (const float*)d_in[11];
    const float* dconvb = (const float*)d_in[12];
    const float* outW   = (const float*)d_in[13];
    const float* outb   = (const float*)d_in[14];
    const float* Wo1    = (const float*)d_in[15];
    const float* bo1    = (const float*)d_in[16];
    const float* Wo2    = (const float*)d_in[17];
    const float* bo2    = (const float*)d_in[18];
    float* out = (float*)d_out;

    // workspace layout (f32 x ping/pong + skip, dp, bf16 weights) ~403 MB
    const size_t NX = (size_t)B_ * L_ * C_;
    float* xA   = (float*)d_ws;
    float* xB   = xA + NX;
    float* skip = xB + NX;
    float* dp   = skip + NX;
    u16* convWb = (u16*)(dp + (size_t)LAYERS_ * B_ * C_);
    u16* outWb  = convWb + (size_t)LAYERS_ * 3 * 128 * C_;
    u16* Wo1b   = outWb  + (size_t)LAYERS_ * 128 * C_;

    embed_dp_kernel<<<B_, 256, 0, stream>>>(step, table, W1, b1, W2, b2, diffW, diffb, dp);

    {
        int total = LAYERS_ * 3 * 128 * C_ + LAYERS_ * 128 * C_ + C_ * C_;
        prep_weights_kernel<<<(total + 255) / 256, 256, 0, stream>>>(
            dconvW, outW, Wo1, convWb, outWb, Wo1b);
    }

    input_proj_kernel<<<(int)(NX / 256), 256, 0, stream>>>(audio, Win, bin, xA, skip);

    const int ntiles = B_ * (L_ / 64);
    for (int i = 0; i < LAYERS_; ++i) {
        int d = 1 << (i % 10);
        const float* xi = (i & 1) ? xB : xA;
        float*       xo = (i & 1) ? xA : xB;
        layer_kernel<<<ntiles, 256, 0, stream>>>(
            xi, xo, skip, dp + (size_t)i * B_ * C_,
            convWb + (size_t)i * 3 * 128 * C_, outWb + (size_t)i * 128 * C_,
            dconvb + (size_t)i * 128, outb + (size_t)i * 128, d);
    }

    final_kernel<<<ntiles, 256, 0, stream>>>(skip, Wo1b, bo1, Wo2, bo2, out);
}
